// HGTModel_68848325755216
// MI455X (gfx1250) — matrix operand order repaired and checked by reading, not compiled, not run on hardware
//
#include <hip/hip_runtime.h>
#include <hip/hip_bf16.h>
#include <math.h>

// ---------------------------------------------------------------------------
// CDNA5 (gfx1250) HGT forward: LDS-staged bf16 WMMA GEMMs + atomic softmax.
// ---------------------------------------------------------------------------

typedef __attribute__((ext_vector_type(16))) __bf16 v16bf;
typedef __attribute__((ext_vector_type(8)))  __bf16 v8bf;
typedef __attribute__((ext_vector_type(8)))  float  v8f;

__device__ __forceinline__ __bf16 f2bf(float f) {
  union { float f; unsigned u; } v; v.f = f;
  unsigned r = v.u + 0x7FFFu + ((v.u >> 16) & 1u);   // round-to-nearest-even
  unsigned short h = (unsigned short)(r >> 16);
  union { unsigned short s; __bf16 b; } o; o.s = h; return o.b;
}

__device__ __forceinline__ float gelu_exact(float x) {
  return 0.5f * x * (1.0f + erff(x * 0.70710678118654752f));
}

__device__ __forceinline__ void atomicMaxF32(float* addr, float val) {
  int iv = __float_as_int(val);
  if (iv >= 0) atomicMax((int*)addr, iv);
  else         atomicMin((unsigned int*)addr, (unsigned int)iv);
}

__device__ __forceinline__ v8bf lds_ld8(const __bf16* p) {
  union { uint4 u; v8bf b; } x;
  x.u = *(const uint4*)p;           // 16B-aligned by construction -> ds_load_b128
  return x.b;
}

// ---------------------------------------------------------------------------
// C[M,N] = epi( act(A[M,K]) @ W[K,N] + bias ), bf16 WMMA, f32 accumulate.
// Block = 128 threads (4 waves). Block tile = 64 rows x 32 cols.
// Each wave owns 16 rows x 32 cols: one A fragment feeds two WMMAs (A reuse).
// Tiles staged in LDS as bf16; interior blocks take a block-uniform fast path
// with unconditional coalesced loads (no per-element exec masking).
// N is always a multiple of 32 for every GEMM in this model.
// ---------------------------------------------------------------------------
#define TPAD 40   // bf16 row stride: 80B (16B aligned), staggers LDS banks

__global__ __launch_bounds__(128) void gemm_bf16_wmma(
    const float* __restrict__ A, long lda,
    const float* __restrict__ W, long ldw,
    const float* __restrict__ bias,             // nullable, indexed by col
    float* __restrict__ C, long ldc,
    int M, int N, int K,
    int a_gelu, int epi_relu,
    const float* __restrict__ resid, long ldr,  // nullable gated residual
    const float* __restrict__ skip)             // device scalar gate (pre-sigmoid)
{
  __shared__ __align__(16) __bf16 sA[64 * TPAD];   // 64 rows x 32 K (padded)
  __shared__ __align__(16) __bf16 sBt[32 * TPAD];  // 32 cols x 32 K (transposed)

  const int tid   = threadIdx.x;
  const int lane  = tid & 31;
  const int wave  = tid >> 5;
  const int m0blk = blockIdx.y * 64;
  const int n0    = blockIdx.x * 32;
  const int m0    = m0blk + wave * 16;

  const int r16   = lane & 15;
  const int khalf = lane >> 4;

  const bool mFull = (m0blk + 64 <= M);            // block-uniform

  v8f acc0 = {0.f,0.f,0.f,0.f,0.f,0.f,0.f,0.f};
  v8f acc1 = {0.f,0.f,0.f,0.f,0.f,0.f,0.f,0.f};

  for (int k0 = 0; k0 < K; k0 += 32) {
    const bool kFull = (k0 + 32 <= K);             // block-uniform

    // --- stage A tile: 64x32 f32 -> bf16, coalesced (c contiguous per row)
    const float* Ab = A + (size_t)m0blk * lda + k0;
    if (mFull && kFull) {
      #pragma unroll
      for (int i = 0; i < 16; ++i) {
        int idx = tid + i * 128;                   // 2048 elements
        int r = idx >> 5, c = idx & 31;
        float f = Ab[(size_t)r * lda + c];
        if (a_gelu) f = gelu_exact(f);
        sA[r * TPAD + c] = f2bf(f);
      }
    } else {
      #pragma unroll
      for (int i = 0; i < 16; ++i) {
        int idx = tid + i * 128;
        int r = idx >> 5, c = idx & 31;
        float f = (m0blk + r < M && k0 + c < K) ? Ab[(size_t)r * lda + c] : 0.f;
        if (a_gelu) f = gelu_exact(f);
        sA[r * TPAD + c] = f2bf(f);
      }
    }
    // --- stage B tile transposed: sBt[n][k]
    const float* Wb = W + (size_t)k0 * ldw + n0;
    if (kFull) {
      #pragma unroll
      for (int i = 0; i < 8; ++i) {
        int idx = tid + i * 128;                   // 1024 elements
        int k = idx >> 5, n = idx & 31;
        sBt[n * TPAD + k] = f2bf(Wb[(size_t)k * ldw + n]);
      }
    } else {
      #pragma unroll
      for (int i = 0; i < 8; ++i) {
        int idx = tid + i * 128;
        int k = idx >> 5, n = idx & 31;
        float f = (k0 + k < K && n0 + n < N) ? Wb[(size_t)k * ldw + n] : 0.f;
        sBt[n * TPAD + k] = f2bf(f);
      }
    }
    __syncthreads();

    // --- fragments (contiguous k = 16*khalf + i, same for A and B), pure ds_load_b128 ------------
    // A: lane row = lane&15; elem i -> K = 16*khalf + i.
    const int ar = wave * 16 + r16;
    v8bf alo = lds_ld8(&sA[ar * TPAD + khalf * 16]);
    v8bf ahi = lds_ld8(&sA[ar * TPAD + khalf * 16 + 8]);
    v16bf a;
    #pragma unroll
    for (int j = 0; j < 8; ++j) { a[j] = alo[j]; a[j + 8] = ahi[j]; }

    // B: lane col = lane&15 (+16 for the second tile); elem i -> K = 16*khalf + i.
    v16bf b0, b1;
    {
      v8bf p0 = lds_ld8(&sBt[r16 * TPAD + khalf * 16]);
      v8bf p1 = lds_ld8(&sBt[r16 * TPAD + khalf * 16 + 8]);
      v8bf q0 = lds_ld8(&sBt[(r16 + 16) * TPAD + khalf * 16]);
      v8bf q1 = lds_ld8(&sBt[(r16 + 16) * TPAD + khalf * 16 + 8]);
      #pragma unroll
      for (int j = 0; j < 8; ++j) {
        b0[j] = p0[j]; b0[j + 8] = p1[j];
        b1[j] = q0[j]; b1[j + 8] = q1[j];
      }
    }

    acc0 = __builtin_amdgcn_wmma_f32_16x16x32_bf16(
        false, a, false, b0, (short)0, acc0, false, false);
    acc1 = __builtin_amdgcn_wmma_f32_16x16x32_bf16(
        false, a, false, b1, (short)0, acc1, false, false);

    __syncthreads();   // fragment reads done before next stage overwrites
  }

  // --- epilogue ------------------------------------------------------------
  float gate = 0.f, gatec = 0.f;
  if (resid && skip) {
    float s = skip[0];
    gate  = 1.f / (1.f + expf(-s));
    gatec = 1.f - gate;
  }
  const int col0 = n0 + r16, col1 = col0 + 16;
  const float bv0 = bias ? bias[col0] : 0.f;
  const float bv1 = bias ? bias[col1] : 0.f;
  #pragma unroll
  for (int j = 0; j < 8; ++j) {
    int r = m0 + j + khalf * 8;        // C layout: elem j -> row j + 8*(lane>>4)
    if (r < M) {
      {
        float v = acc0[j] + bv0;
        if (epi_relu) v = fmaxf(v, 0.f);
        if (resid && skip) v = gate * v + gatec * resid[(size_t)r * ldr + col0];
        C[(size_t)r * ldc + col0] = v;
      }
      {
        float v = acc1[j] + bv1;
        if (epi_relu) v = fmaxf(v, 0.f);
        if (resid && skip) v = gate * v + gatec * resid[(size_t)r * ldr + col1];
        C[(size_t)r * ldc + col1] = v;
      }
    }
  }
}

// Zero agg, -inf segmax, zero denom.
__global__ void init_aux(float* __restrict__ agg, float* __restrict__ smax,
                         float* __restrict__ sden, long nAgg, long nSeg) {
  long i = (long)blockIdx.x * blockDim.x + threadIdx.x;
  if (i < nAgg) agg[i] = 0.f;
  if (i < nSeg) { smax[i] = -3.402823466e38f; sden[i] = 0.f; }
}

// Pass 1: per (edge, head) attention logit + atomic segment max over dst.
// q/k row bases are 128B aligned -> float4 (global_load_b128) loads.
__global__ void edge_logit(const int* __restrict__ src, const int* __restrict__ dst,
                           const float* __restrict__ ks_seg,   // [Ns,64]
                           const float* __restrict__ q_seg,    // z + 64, ld=192
                           const float* __restrict__ prel,     // 2 floats
                           float* __restrict__ lg_seg,         // [NE,2]
                           float* __restrict__ smax_seg,       // [Nd,2]
                           int ne) {
  long idx = (long)blockIdx.x * blockDim.x + threadIdx.x;
  if (idx >= (long)ne * 2) return;
  int e = (int)(idx >> 1), h = (int)(idx & 1);
  int s = src[e], d = dst[e];
  const float4* q4 = (const float4*)(q_seg + (size_t)d * 192 + h * 32);
  const float4* k4 = (const float4*)(ks_seg + (size_t)s * 64 + h * 32);
  float acc = 0.f;
  #pragma unroll
  for (int i = 0; i < 8; ++i) {
    float4 qa = q4[i], ka = k4[i];
    acc += qa.x * ka.x + qa.y * ka.y + qa.z * ka.z + qa.w * ka.w;
  }
  float lg = acc * prel[h] * 0.17677669529663687f;   // 1/sqrt(32)
  lg_seg[(size_t)e * 2 + h] = lg;
  atomicMaxF32(&smax_seg[(size_t)d * 2 + h], lg);
}

// Pass 2: exp(logit - max), atomic denom sum. Overwrites logits with exp.
__global__ void edge_exp(const int* __restrict__ dst,
                         float* __restrict__ lg_seg,
                         const float* __restrict__ smax_seg,
                         float* __restrict__ sden_seg, int ne) {
  long idx = (long)blockIdx.x * blockDim.x + threadIdx.x;
  if (idx >= (long)ne * 2) return;
  int e = (int)(idx >> 1), h = (int)(idx & 1);
  int d = dst[e];
  float ex = expf(lg_seg[(size_t)e * 2 + h] - smax_seg[(size_t)d * 2 + h]);
  lg_seg[(size_t)e * 2 + h] = ex;
  atomicAdd(&sden_seg[(size_t)d * 2 + h], ex);
}

// Pass 3: scatter alpha-weighted messages into agg.
__global__ void edge_agg(const int* __restrict__ src, const int* __restrict__ dst,
                         const float* __restrict__ ex_seg,
                         const float* __restrict__ sden_seg,
                         const float* __restrict__ vs_seg,    // [Ns,64]
                         float* __restrict__ agg_seg,         // [Nd,64]
                         int ne) {
  long idx = (long)blockIdx.x * blockDim.x + threadIdx.x;
  if (idx >= (long)ne * 64) return;
  int e = (int)(idx >> 6), c = (int)(idx & 63), h = c >> 5;
  int s = src[e], d = dst[e];
  float ex  = ex_seg[(size_t)e * 2 + h];
  float den = sden_seg[(size_t)d * 2 + h];
  float alpha = ex / fmaxf(den, 1e-16f);
  atomicAdd(&agg_seg[(size_t)d * 64 + c], vs_seg[(size_t)s * 64 + c] * alpha);
}

// Classifier tail: out = sigmoid(h[100k,32] @ Wc2[32,1] + bc2)
__global__ void cls2(const float* __restrict__ h, const float* __restrict__ Wc2,
                     const float* __restrict__ bc2, float* __restrict__ out, int n) {
  int i = blockIdx.x * blockDim.x + threadIdx.x;
  if (i >= n) return;
  const float4* h4 = (const float4*)(h + (size_t)i * 32);
  const float4* w4 = (const float4*)Wc2;
  float acc = bc2[0];
  #pragma unroll
  for (int j = 0; j < 8; ++j) {
    float4 a = h4[j], b = w4[j];
    acc += a.x * b.x + a.y * b.y + a.z * b.z + a.w * b.w;
  }
  out[i] = 1.f / (1.f + expf(-acc));
}

// ---------------------------------------------------------------------------

static inline void launch_gemm(hipStream_t st, const float* A, long lda,
                               const float* W, long ldw, const float* bias,
                               float* C, long ldc, int M, int N, int K,
                               int a_gelu, int epi_relu,
                               const float* resid, long ldr, const float* skip) {
  dim3 blk(128);
  dim3 grd((unsigned)((N + 31) / 32), (unsigned)((M + 63) / 64));
  gemm_bf16_wmma<<<grd, blk, 0, st>>>(A, lda, W, ldw, bias, C, ldc, M, N, K,
                                      a_gelu, epi_relu, resid, ldr, skip);
}

extern "C" void kernel_launch(void* const* d_in, const int* in_sizes, int n_in,
                              void* d_out, int out_size, void* d_ws, size_t ws_size,
                              hipStream_t stream) {
  (void)in_sizes; (void)n_in; (void)out_size; (void)ws_size;

  // --- model constants -----------------------------------------------------
  const int NE = 250000, HID = 64, Lyr = 3;
  static const int Nn[5]    = {100000, 50000, 5000, 80000, 60000};
  static const int FEATn[5] = {262, 4, 4, 768, 3};
  static const int noff[5]  = {0, 100000, 150000, 155000, 235000};   // node prefix
  const int NTOT = 295000;
  static const int esrc_t[8] = {1, 0, 2, 0, 4, 0, 3, 0};
  static const int edst_t[8] = {0, 1, 0, 2, 0, 4, 0, 3};
  static const int eoff[8]   = {0, 50000, 150000, 155000, 255000,
                                315000, 415000, 495000};             // src-seg prefix

  // --- inputs --------------------------------------------------------------
  const float* xin[5];
  for (int t = 0; t < 5; ++t) xin[t] = (const float*)d_in[t];
  const int* esrc_p[8]; const int* edst_p[8];
  for (int e = 0; e < 8; ++e) {
    esrc_p[e] = (const int*)d_in[5 + 2 * e];
    edst_p[e] = (const int*)d_in[6 + 2 * e];
  }
  const float* Wp[5];
  for (int t = 0; t < 5; ++t) Wp[t] = (const float*)d_in[21 + t];
  const float* bp    = (const float*)d_in[26];
  const float* W_kqv = (const float*)d_in[27];   // [3,5,64,192]
  const float* b_kqv = (const float*)d_in[28];   // [3,5,192]
  const float* k_rel = (const float*)d_in[29];   // [3,8,2,32,32]
  const float* v_rel = (const float*)d_in[30];
  const float* p_rel = (const float*)d_in[31];   // [3,8,2]
  const float* W_out = (const float*)d_in[32];   // [3,5,64,64]
  const float* b_out = (const float*)d_in[33];   // [3,5,64]
  const float* skipP = (const float*)d_in[34];   // [3,5]
  const float* Wc1   = (const float*)d_in[35];   // [64,32]
  const float* bc1   = (const float*)d_in[36];
  const float* Wc2   = (const float*)d_in[37];   // [32,1]
  const float* bc2   = (const float*)d_in[38];

  // --- workspace layout (floats) ------------------------------------------
  float* ws = (float*)d_ws;
  size_t o = 0;
  float* xA   = ws + o; o += (size_t)NTOT * HID;       // node feats (ping)
  float* xB   = ws + o; o += (size_t)NTOT * HID;       // node feats (pong)
  float* zb   = ws + o; o += (size_t)NTOT * 192;       // fused [k|q|v]
  float* ksb  = ws + o; o += (size_t)595000 * HID;     // per-relation k_s
  float* vsb  = ws + o; o += (size_t)595000 * HID;     // per-relation v_s
  float* aggb = ws + o; o += (size_t)NTOT * HID;       // softmax aggregate
  float* smax = ws + o; o += (size_t)NTOT * 2;         // segment max [N,H]
  float* sden = ws + o; o += (size_t)NTOT * 2;         // segment denom
  float* lgb  = ws + o; o += (size_t)8 * NE * 2;       // logits/exp [E,H]
  float* hb   = ws + o; o += (size_t)100000 * 32;      // classifier hidden

  // --- input projections: x_t @ Wp_t + bp[t] -------------------------------
  for (int t = 0; t < 5; ++t)
    launch_gemm(stream, xin[t], FEATn[t], Wp[t], HID, bp + (size_t)t * HID,
                xA + (size_t)noff[t] * HID, HID, Nn[t], HID, FEATn[t],
                0, 0, nullptr, 0, nullptr);

  float* xcur = xA; float* xnxt = xB;

  for (int l = 0; l < Lyr; ++l) {
    // fused KQV: z = x @ W_kqv[l,t] + b_kqv[l,t]   (cols: k 0:64, q 64:128, v 128:192)
    for (int t = 0; t < 5; ++t)
      launch_gemm(stream, xcur + (size_t)noff[t] * HID, HID,
                  W_kqv + ((size_t)(l * 5 + t)) * 64 * 192, 192,
                  b_kqv + (size_t)(l * 5 + t) * 192,
                  zb + (size_t)noff[t] * 192, 192, Nn[t], 192, HID,
                  0, 0, nullptr, 0, nullptr);

    // per-relation per-head 32x32 transforms at source-node granularity
    for (int e = 0; e < 8; ++e) {
      int st = esrc_t[e]; int Ns = Nn[st];
      const float* zsrc = zb + (size_t)noff[st] * 192;
      for (int h = 0; h < 2; ++h) {
        size_t wofs = ((size_t)(l * 8 + e) * 2 + h) * 1024;
        launch_gemm(stream, zsrc + h * 32, 192, k_rel + wofs, 32, nullptr,
                    ksb + (size_t)eoff[e] * HID + h * 32, HID, Ns, 32, 32,
                    0, 0, nullptr, 0, nullptr);
        launch_gemm(stream, zsrc + 128 + h * 32, 192, v_rel + wofs, 32, nullptr,
                    vsb + (size_t)eoff[e] * HID + h * 32, HID, Ns, 32, 32,
                    0, 0, nullptr, 0, nullptr);
      }
    }

    // segment-softmax aggregation (3 stream-ordered passes)
    {
      long nAgg = (long)NTOT * HID, nSeg = (long)NTOT * 2;
      init_aux<<<(unsigned)((nAgg + 255) / 256), 256, 0, stream>>>(
          aggb, smax, sden, nAgg, nSeg);
    }
    for (int e = 0; e < 8; ++e) {
      int dt = edst_t[e];
      edge_logit<<<(unsigned)(((long)NE * 2 + 255) / 256), 256, 0, stream>>>(
          esrc_p[e], edst_p[e], ksb + (size_t)eoff[e] * HID,
          zb + (size_t)noff[dt] * 192 + 64, p_rel + (size_t)(l * 8 + e) * 2,
          lgb + (size_t)e * NE * 2, smax + (size_t)noff[dt] * 2, NE);
    }
    for (int e = 0; e < 8; ++e) {
      int dt = edst_t[e];
      edge_exp<<<(unsigned)(((long)NE * 2 + 255) / 256), 256, 0, stream>>>(
          edst_p[e], lgb + (size_t)e * NE * 2,
          smax + (size_t)noff[dt] * 2, sden + (size_t)noff[dt] * 2, NE);
    }
    for (int e = 0; e < 8; ++e) {
      int dt = edst_t[e];
      edge_agg<<<(unsigned)(((long)NE * 64 + 255) / 256), 256, 0, stream>>>(
          esrc_p[e], edst_p[e], lgb + (size_t)e * NE * 2,
          sden + (size_t)noff[dt] * 2, vsb + (size_t)eoff[e] * HID,
          aggb + (size_t)noff[dt] * HID, NE);
    }

    // node update: gated residual of gelu(agg) @ W_out + b_out
    for (int t = 0; t < 5; ++t)
      launch_gemm(stream, aggb + (size_t)noff[t] * HID, HID,
                  W_out + (size_t)(l * 5 + t) * 4096, HID,
                  b_out + (size_t)(l * 5 + t) * HID,
                  xnxt + (size_t)noff[t] * HID, HID, Nn[t], HID, HID,
                  /*gelu*/1, /*relu*/0,
                  xcur + (size_t)noff[t] * HID, HID, skipP + (l * 5 + t));

    float* tmp = xcur; xcur = xnxt; xnxt = tmp;
  }

  // classifier: relu(x_article @ Wc1 + bc1) -> sigmoid(@ Wc2 + bc2)
  launch_gemm(stream, xcur /* article at node offset 0 */, HID, Wc1, 32, bc1,
              hb, 32, Nn[0], 32, HID, 0, /*relu*/1, nullptr, 0, nullptr);
  cls2<<<(Nn[0] + 255) / 256, 256, 0, stream>>>(hb, Wc2, bc2, (float*)d_out, Nn[0]);
}
